// MultiheadAttention_20581483282551
// MI455X (gfx1250) — matrix operand order repaired and checked by reading, not compiled, not run on hardware
//
#include <hip/hip_runtime.h>
#include <hip/hip_bf16.h>

// ---- problem constants (match reference) ----
#define BB  2
#define SS  2048
#define DD  1024
#define HH  16
#define DHH 64

typedef __bf16 v16bf __attribute__((ext_vector_type(16)));
typedef __bf16 v8bf  __attribute__((ext_vector_type(8)));
typedef float  v8f   __attribute__((ext_vector_type(8)));
typedef int    v4i   __attribute__((vector_size(4 * sizeof(int))));

union AFrag { v16bf v; v8bf h[2]; };

static __device__ __forceinline__ v8f wmma_bf16(v16bf a, v16bf b, v8f c) {
  // D = A(16x32 bf16) * B(32x16 bf16) + C(16x16 f32)
  return __builtin_amdgcn_wmma_f32_16x16x32_bf16(false, a, false, b, (short)0, c, false, false);
}

// pack two bf16 into one dword (so transposed LDS staging uses ds_store_b32)
static __device__ __forceinline__ unsigned pack_bf16(__bf16 lo, __bf16 hi) {
  union { unsigned u; __bf16 b[2]; } p;
  p.b[0] = lo; p.b[1] = hi;
  return p.u;
}

// ---- async global->LDS (CDNA5 ASYNCcnt path), guarded with fallback ----
#if defined(__has_builtin)
#if __has_builtin(__builtin_amdgcn_global_load_async_to_lds_b128) && \
    __has_builtin(__builtin_amdgcn_s_wait_asynccnt)
#define USE_ASYNC_LDS 1
#endif
#endif
#ifndef USE_ASYNC_LDS
#define USE_ASYNC_LDS 0
#endif

static __device__ __forceinline__ void async_ld_b128(const void* g, void* l) {
#if USE_ASYNC_LDS
  // builtin signature (from clang diagnostic): (int4 __device__*, int4 __shared__*, imm offset, cpol)
  __builtin_amdgcn_global_load_async_to_lds_b128(
      (__attribute__((address_space(1))) v4i*)g,
      (__attribute__((address_space(3))) v4i*)l, 0, 0);
#else
  *(int4*)l = *(const int4*)g;
#endif
}
static __device__ __forceinline__ void wait_async() {
#if USE_ASYNC_LDS
  __builtin_amdgcn_s_wait_asynccnt(0);
#endif
}

// =====================================================================
// Kernel 1: projection GEMM.  Y[b,h,s,dh] = sum_k X[b,s,k] * W[k, h*64+dh]
// X fp32 [B*S, D] row-major, W fp32 [D, D] ([in,out]), output bf16 in
// head-split layout [B,H,S,DH].  Block tile 64x64, K-chunk 32 through LDS.
// 256 threads = 8 waves; wave = (mt in 0..3) x (ngroup in 0..1, 2 n-tiles).
// =====================================================================
__global__ __launch_bounds__(256) void gemm_proj_kernel(
    const float* __restrict__ X, const float* __restrict__ W,
    __bf16* __restrict__ Yh) {
  __shared__ __attribute__((aligned(32))) __bf16 As[64][32];   // [m][k]
  __shared__ __attribute__((aligned(32))) __bf16 Bs[64][32];   // [n][k]

  const int tid  = threadIdx.x;
  const int lane = tid & 31;
  const int wave = tid >> 5;
  const int hh   = lane >> 4;   // half-wave 0/1
  const int ln   = lane & 15;
  const int m0   = blockIdx.x * 64;
  const int n0   = blockIdx.y * 64;
  const int mt   = wave & 3;    // m-tile 0..3
  const int ng   = wave >> 2;   // n-group 0/1 -> n-tiles {2ng, 2ng+1}

  const int arow = tid >> 2;          // 0..63
  const int acol = (tid & 3) * 8;     // 0,8,16,24
  const int k2   = tid >> 4;          // 0..15 -> k rows {2k2, 2k2+1}
  const int n4   = (tid & 15) * 4;    // n offset 0..60

  v8f acc[2] = {v8f{}, v8f{}};

  for (int kk = 0; kk < DD; kk += 32) {
    // stage A tile (fp32 -> bf16), pairwise -> v_cvt_pk_bf16_f32 + b128 store
    const float* ap = X + (size_t)(m0 + arow) * DD + kk + acol;
    float4 a0 = *(const float4*)ap;
    float4 a1 = *(const float4*)(ap + 4);
    As[arow][acol + 0] = (__bf16)a0.x; As[arow][acol + 1] = (__bf16)a0.y;
    As[arow][acol + 2] = (__bf16)a0.z; As[arow][acol + 3] = (__bf16)a0.w;
    As[arow][acol + 4] = (__bf16)a1.x; As[arow][acol + 5] = (__bf16)a1.y;
    As[arow][acol + 6] = (__bf16)a1.z; As[arow][acol + 7] = (__bf16)a1.w;

    // stage B tile transposed: Bs[n][k] = W[k][n].  Each thread owns k rows
    // {2k2, 2k2+1} x 4 n's so each store is a packed bf16 pair (ds_store_b32).
    const float* bp0 = W + (size_t)(kk + 2 * k2 + 0) * DD + n0 + n4;
    const float* bp1 = W + (size_t)(kk + 2 * k2 + 1) * DD + n0 + n4;
    float4 w0 = *(const float4*)bp0;
    float4 w1 = *(const float4*)bp1;
    const float* w0f = (const float*)&w0;
    const float* w1f = (const float*)&w1;
#pragma unroll
    for (int j = 0; j < 4; ++j) {
      *(unsigned*)&Bs[n4 + j][2 * k2] =
          pack_bf16((__bf16)w0f[j], (__bf16)w1f[j]);
    }

    // prefetch next K-chunk
    if (kk + 32 < DD) {
      __builtin_prefetch(ap + 32, 0, 1);
      __builtin_prefetch(bp0 + 32 * DD, 0, 1);
    }
    __syncthreads();

    // A fragment: lane(hh, m=ln): K = 16hh+e, contiguous
    AFrag a;
    a.h[0] = *(const v8bf*)&As[mt * 16 + ln][16 * hh];
    a.h[1] = *(const v8bf*)&As[mt * 16 + ln][16 * hh + 8];
    // B fragments: lane(hh, n=ln): K = 16hh+e, contiguous
    v16bf bf0 = *(const v16bf*)&Bs[(2 * ng + 0) * 16 + ln][16 * hh];
    v16bf bf1 = *(const v16bf*)&Bs[(2 * ng + 1) * 16 + ln][16 * hh];
    acc[0] = wmma_bf16(a.v, bf0, acc[0]);
    acc[1] = wmma_bf16(a.v, bf1, acc[1]);
    __syncthreads();
  }

  // epilogue: scatter into head-split bf16 [B,H,S,DH]
#pragma unroll
  for (int t = 0; t < 2; ++t) {
    const int gn = n0 + (2 * ng + t) * 16 + ln;
    const int head = gn >> 6;     // /64
    const int dh = gn & 63;
#pragma unroll
    for (int r = 0; r < 8; ++r) {
      const int gm = m0 + mt * 16 + r + 8 * hh;
      const int b = gm >> 11;     // /2048
      const int s = gm & 2047;
      Yh[(((size_t)b * HH + head) * SS + s) * DHH + dh] = (__bf16)acc[t][r];
    }
  }
}

// =====================================================================
// Kernel 2: flash attention per (b,h).  Unscaled QK^T (faithful to ref),
// online softmax, PV accumulation.  Block = 64 query rows, 4 waves; K/V
// consumed in 32-row blocks staged in LDS (K via async-to-LDS copy when
// available; V transposed with packed b32 stores).
// =====================================================================
__global__ __launch_bounds__(128) void flash_attn_kernel(
    const __bf16* __restrict__ Qh, const __bf16* __restrict__ Kh,
    const __bf16* __restrict__ Vh, __bf16* __restrict__ Ocat) {
  __shared__ __attribute__((aligned(32))) __bf16 Ks[32][64];      // [krow][feat]
  __shared__ __attribute__((aligned(32))) __bf16 Vt[64][32];      // [dh][krow]
  __shared__ __attribute__((aligned(32))) __bf16 Pt[4][16][32];   // per-wave P

  const int tid  = threadIdx.x;
  const int lane = tid & 31;
  const int wave = tid >> 5;    // 0..3
  const int hh   = lane >> 4;
  const int ln   = lane & 15;
  const int bh   = blockIdx.y;             // b*H + head
  const int b    = bh >> 4;
  const int head = bh & 15;
  const size_t base = (size_t)bh * SS * DHH;
  const int qm0 = blockIdx.x * 64;
  const int qrow = qm0 + wave * 16 + ln;   // A-fragment row

  // Q fragments (resident for whole kernel): features kk*32 + 16hh + e (contiguous)
  AFrag aq[2];
  const __bf16* qb = Qh + base + (size_t)qrow * DHH;
#pragma unroll
  for (int kk = 0; kk < 2; ++kk) {
    aq[kk].h[0] = *(const v8bf*)(qb + kk * 32 + 16 * hh);
    aq[kk].h[1] = *(const v8bf*)(qb + kk * 32 + 16 * hh + 8);
  }

  v8f accO[4] = {v8f{}, v8f{}, v8f{}, v8f{}};
  float mrow[8], lrow[8];
#pragma unroll
  for (int r = 0; r < 8; ++r) { mrow[r] = -3.0e38f; lrow[r] = 0.0f; }

  const int srow = tid >> 2;          // 0..31 (K block row)
  const int scol = (tid & 3) * 16;    // 0/16/32/48 (feature chunk)
  const int vk2  = tid >> 3;          // 0..15 -> V rows {2vk2, 2vk2+1}
  const int vdh  = (tid & 7) * 8;     // dh offset 0..56

  for (int jj = 0; jj < SS / 32; ++jj) {
    // stage K rows [jj*32, jj*32+32): pure bf16 copy -> async-to-LDS
    const __bf16* kp = Kh + base + (size_t)(jj * 32 + srow) * DHH + scol;
    async_ld_b128(kp, &Ks[srow][scol]);
    async_ld_b128(kp + 8, &Ks[srow][scol + 8]);

    // stage V transposed: Vt[dh][krow], packed bf16 pairs along k
    const __bf16* vp0 = Vh + base + (size_t)(jj * 32 + 2 * vk2 + 0) * DHH + vdh;
    const __bf16* vp1 = Vh + base + (size_t)(jj * 32 + 2 * vk2 + 1) * DHH + vdh;
    v8bf va = *(const v8bf*)vp0;
    v8bf vb = *(const v8bf*)vp1;
#pragma unroll
    for (int j = 0; j < 8; ++j)
      *(unsigned*)&Vt[vdh + j][2 * vk2] = pack_bf16(va[j], vb[j]);

    if (jj + 1 < SS / 32) {
      __builtin_prefetch(kp + 32 * DHH, 0, 1);
      __builtin_prefetch(vp0 + 32 * DHH, 0, 1);
    }
    wait_async();
    __syncthreads();

    // scores: S[m][c] = sum_f Q[m][f] * K[jj*32 + c][f], two 16-col tiles
    v16bf bk00 = *(const v16bf*)&Ks[ln][16 * hh];
    v16bf bk01 = *(const v16bf*)&Ks[ln][32 + 16 * hh];
    v16bf bk10 = *(const v16bf*)&Ks[16 + ln][16 * hh];
    v16bf bk11 = *(const v16bf*)&Ks[16 + ln][32 + 16 * hh];
    v8f z = {};
    v8f s0 = wmma_bf16(aq[0].v, bk00, z);
    s0 = wmma_bf16(aq[1].v, bk01, s0);
    v8f s1 = wmma_bf16(aq[0].v, bk10, z);
    s1 = wmma_bf16(aq[1].v, bk11, s1);

    // online softmax (row m = r + 8*hh lives across 16 lanes of this half)
    float p0[8], p1[8], sc[8];
#pragma unroll
    for (int r = 0; r < 8; ++r) {
      float v0 = s0[r], v1 = s1[r];
      float rm = fmaxf(v0, v1);
#pragma unroll
      for (int mk = 1; mk <= 8; mk <<= 1) rm = fmaxf(rm, __shfl_xor(rm, mk, 32));
      float mnew = fmaxf(mrow[r], rm);
      float scale = __expf(mrow[r] - mnew);
      float e0 = __expf(v0 - mnew);
      float e1 = __expf(v1 - mnew);
      float rs = e0 + e1;
#pragma unroll
      for (int mk = 1; mk <= 8; mk <<= 1) rs += __shfl_xor(rs, mk, 32);
      lrow[r] = lrow[r] * scale + rs;
      mrow[r] = mnew;
      sc[r] = scale; p0[r] = e0; p1[r] = e1;
    }
#pragma unroll
    for (int t = 0; t < 4; ++t)
#pragma unroll
      for (int r = 0; r < 8; ++r) accO[t][r] = accO[t][r] * sc[r];

    // C-layout -> A-layout for P via per-wave LDS tile
#pragma unroll
    for (int r = 0; r < 8; ++r) {
      Pt[wave][r + 8 * hh][ln]      = (__bf16)p0[r];
      Pt[wave][r + 8 * hh][16 + ln] = (__bf16)p1[r];
    }
    AFrag ap;
    ap.h[0] = *(const v8bf*)&Pt[wave][ln][16 * hh];
    ap.h[1] = *(const v8bf*)&Pt[wave][ln][16 * hh + 8];

    // O += P(16x32) @ V(32x64): B[k][n] = Vt[t*16+n][k]
#pragma unroll
    for (int t = 0; t < 4; ++t) {
      v16bf bv = *(const v16bf*)&Vt[t * 16 + ln][16 * hh];
      accO[t] = wmma_bf16(ap.v, bv, accO[t]);
    }
    __syncthreads();
  }

  // epilogue: normalize and store concat-head bf16 [B,S,D]
#pragma unroll
  for (int t = 0; t < 4; ++t) {
#pragma unroll
    for (int r = 0; r < 8; ++r) {
      const int s = qm0 + wave * 16 + r + 8 * hh;
      const float o = accO[t][r] / lrow[r];
      Ocat[((size_t)b * SS + s) * DD + head * 64 + t * 16 + ln] = (__bf16)o;
    }
  }
}

// =====================================================================
// Kernel 3: output projection.  C[m][n] = sum_k A_bf16[m][k] * Wo[k][n],
// fp32 output.  A staging is a pure bf16 copy -> async-to-LDS.
// =====================================================================
__global__ __launch_bounds__(256) void gemm_out_kernel(
    const __bf16* __restrict__ A, const float* __restrict__ W,
    float* __restrict__ Cout) {
  __shared__ __attribute__((aligned(32))) __bf16 As[64][32];
  __shared__ __attribute__((aligned(32))) __bf16 Bs[64][32];

  const int tid  = threadIdx.x;
  const int lane = tid & 31;
  const int wave = tid >> 5;
  const int hh   = lane >> 4;
  const int ln   = lane & 15;
  const int m0   = blockIdx.x * 64;
  const int n0   = blockIdx.y * 64;
  const int mt   = wave & 3;
  const int ng   = wave >> 2;

  const int arow = tid >> 2;
  const int acol = (tid & 3) * 8;
  const int k2   = tid >> 4;
  const int n4   = (tid & 15) * 4;

  v8f acc[2] = {v8f{}, v8f{}};

  for (int kk = 0; kk < DD; kk += 32) {
    // stage A tile: 16B bf16 copy per thread -> async-to-LDS
    const __bf16* apg = A + (size_t)(m0 + arow) * DD + kk + acol;
    async_ld_b128(apg, &As[arow][acol]);

    // stage B transposed with packed bf16-pair stores
    const float* bp0 = W + (size_t)(kk + 2 * k2 + 0) * DD + n0 + n4;
    const float* bp1 = W + (size_t)(kk + 2 * k2 + 1) * DD + n0 + n4;
    float4 w0 = *(const float4*)bp0;
    float4 w1 = *(const float4*)bp1;
    const float* w0f = (const float*)&w0;
    const float* w1f = (const float*)&w1;
#pragma unroll
    for (int j = 0; j < 4; ++j) {
      *(unsigned*)&Bs[n4 + j][2 * k2] =
          pack_bf16((__bf16)w0f[j], (__bf16)w1f[j]);
    }

    if (kk + 32 < DD) {
      __builtin_prefetch(apg + 32, 0, 1);
      __builtin_prefetch(bp0 + 32 * DD, 0, 1);
    }
    wait_async();
    __syncthreads();

    AFrag a;
    a.h[0] = *(const v8bf*)&As[mt * 16 + ln][16 * hh];
    a.h[1] = *(const v8bf*)&As[mt * 16 + ln][16 * hh + 8];
    v16bf bf0 = *(const v16bf*)&Bs[(2 * ng + 0) * 16 + ln][16 * hh];
    v16bf bf1 = *(const v16bf*)&Bs[(2 * ng + 1) * 16 + ln][16 * hh];
    acc[0] = wmma_bf16(a.v, bf0, acc[0]);
    acc[1] = wmma_bf16(a.v, bf1, acc[1]);
    __syncthreads();
  }

#pragma unroll
  for (int t = 0; t < 2; ++t) {
    const int gn = n0 + (2 * ng + t) * 16 + ln;
#pragma unroll
    for (int r = 0; r < 8; ++r) {
      const int gm = m0 + mt * 16 + r + 8 * hh;
      Cout[(size_t)gm * DD + gn] = acc[t][r];
    }
  }
}

// =====================================================================
extern "C" void kernel_launch(void* const* d_in, const int* in_sizes, int n_in,
                              void* d_out, int out_size, void* d_ws, size_t ws_size,
                              hipStream_t stream) {
  const float* q  = (const float*)d_in[0];
  const float* k  = (const float*)d_in[1];
  const float* v  = (const float*)d_in[2];
  // d_in[3] = valid_length: ignored by the reference attention math
  const float* Wq = (const float*)d_in[4];
  const float* Wk = (const float*)d_in[5];
  const float* Wv = (const float*)d_in[6];
  const float* Wo = (const float*)d_in[7];
  float* out = (float*)d_out;

  const size_t nBHSD = (size_t)BB * HH * SS * DHH;  // 4M elements each
  __bf16* Qh   = (__bf16*)d_ws;
  __bf16* Kh   = Qh + nBHSD;
  __bf16* Vh   = Kh + nBHSD;
  __bf16* Ocat = Vh + nBHSD;                        // [B*S, D] bf16

  const dim3 gGemm((BB * SS) / 64, DD / 64);        // (64, 16)
  gemm_proj_kernel<<<gGemm, 256, 0, stream>>>(q, Wq, Qh);
  gemm_proj_kernel<<<gGemm, 256, 0, stream>>>(k, Wk, Kh);
  gemm_proj_kernel<<<gGemm, 256, 0, stream>>>(v, Wv, Vh);

  flash_attn_kernel<<<dim3(SS / 64, BB * HH), 128, 0, stream>>>(Qh, Kh, Vh, Ocat);

  gemm_out_kernel<<<gGemm, 256, 0, stream>>>(Ocat, Wo, out);
}
